// Darnn_14774687498323
// MI455X (gfx1250) — matrix operand order repaired and checked by reading, not compiled, not run on hardware
//
#include <hip/hip_runtime.h>
#include <hip/hip_bf16.h>
#include <stdint.h>

#define B_SZ 2048
#define T_SZ 64
#define N_SZ 256
#define M_SZ 256

typedef _Float16 h8_t  __attribute__((ext_vector_type(8)));
typedef _Float16 h16_t __attribute__((ext_vector_type(16)));
typedef float    f8_t  __attribute__((ext_vector_type(8)));

__device__ __forceinline__ float sigmf(float x) { return 1.0f / (1.0f + __expf(-x)); }

// ---------------------------------------------------------------------------
// Generic WMMA GEMM:  D[M x N] = A[M x K] * W[N x K]^T (+ bias[N])
// A row-major (lda halves), W row-major (ldw halves, "out x in" like torch).
// Each wave computes a 32x64 tile (2 row-tiles x 4 col-tiles): every B
// fragment is reused by two WMMAs -> 8 WMMAs per 12x16B loads per K-chunk.
// Block = 4 waves -> 128x64 per block. Requires M % 128 == 0, N % 64 == 0,
// K % 32 == 0 (true for all call sites here).
// Fragment layouts match the ISA 16-bit A 16x32 / B 32x16 VGPR mappings, so
// all per-lane fragment loads are contiguous 16-byte chunks.
// ---------------------------------------------------------------------------
__global__ __launch_bounds__(128)
void gemm_wmma_f16(const _Float16* __restrict__ A, int lda,
                   const _Float16* __restrict__ W, int ldw,
                   const float* __restrict__ bias,
                   float* __restrict__ Cf, _Float16* __restrict__ Ch,
                   int ldc, int K)
{
    const int wave  = threadIdx.x >> 5;
    const int lane  = threadIdx.x & 31;
    const int lhalf = lane >> 4;
    const int lmod  = lane & 15;
    const int rowBase = (blockIdx.y * 4 + wave) * 32;
    const int colBase = blockIdx.x * 64;

    const _Float16* aRow0 = A + (size_t)(rowBase + lmod) * lda;
    const _Float16* aRow1 = A + (size_t)(rowBase + 16 + lmod) * lda;
    const _Float16* bRow[4];
#pragma unroll
    for (int ct = 0; ct < 4; ++ct)
        bRow[ct] = W + (size_t)(colBase + ct * 16 + lmod) * ldw;

    f8_t acc0[4] = {};
    f8_t acc1[4] = {};

    for (int k0 = 0; k0 < K; k0 += 32) {
        h8_t a0Lo = *(const h8_t*)(aRow0 + k0 + lhalf * 8);
        h8_t a0Hi = *(const h8_t*)(aRow0 + k0 + 16 + lhalf * 8);
        h16_t af0 = __builtin_shufflevector(a0Lo, a0Hi,
            0, 1, 2, 3, 4, 5, 6, 7, 8, 9, 10, 11, 12, 13, 14, 15);
        h8_t a1Lo = *(const h8_t*)(aRow1 + k0 + lhalf * 8);
        h8_t a1Hi = *(const h8_t*)(aRow1 + k0 + 16 + lhalf * 8);
        h16_t af1 = __builtin_shufflevector(a1Lo, a1Hi,
            0, 1, 2, 3, 4, 5, 6, 7, 8, 9, 10, 11, 12, 13, 14, 15);
        const int kb = k0 + lhalf * 8;
#pragma unroll
        for (int ct = 0; ct < 4; ++ct) {
            h8_t bLo = *(const h8_t*)(bRow[ct] + kb);
            h8_t bHi = *(const h8_t*)(bRow[ct] + kb + 16);
            h16_t bfr = __builtin_shufflevector(bLo, bHi,
                0, 1, 2, 3, 4, 5, 6, 7, 8, 9, 10, 11, 12, 13, 14, 15);
            acc0[ct] = __builtin_amdgcn_wmma_f32_16x16x32_f16(
                false, af0, false, bfr, (short)0, acc0[ct], false, false);
            acc1[ct] = __builtin_amdgcn_wmma_f32_16x16x32_f16(
                false, af1, false, bfr, (short)0, acc1[ct], false, false);
        }
    }

    const int mrow0 = rowBase + lhalf * 8;
    const int mrow1 = rowBase + 16 + lhalf * 8;
#pragma unroll
    for (int ct = 0; ct < 4; ++ct) {
        const int n = colBase + ct * 16 + lmod;
        const float bv = bias ? bias[n] : 0.0f;
#pragma unroll
        for (int i = 0; i < 8; ++i) {
            float v0 = acc0[ct][i] + bv;
            float v1 = acc1[ct][i] + bv;
            size_t off0 = (size_t)(mrow0 + i) * ldc + n;
            size_t off1 = (size_t)(mrow1 + i) * ldc + n;
            if (Cf) { Cf[off0] = v0; Cf[off1] = v1; }
            if (Ch) { Ch[off0] = (_Float16)v0; Ch[off1] = (_Float16)v1; }
        }
    }
}

// ---------------------------------------------------------------------------
// Helpers: conversion / layout / init
// ---------------------------------------------------------------------------
__global__ void k_f32_to_f16(const float* __restrict__ in, _Float16* __restrict__ out, int n)
{
    int i = blockIdx.x * 256 + threadIdx.x;
    if (i < n) out[i] = (_Float16)in[i];
}

__global__ void k_zero_u32(uint32_t* __restrict__ p, int n)
{
    int i = blockIdx.x * 256 + threadIdx.x;
    if (i < n) p[i] = 0u;
}

// Xt[b][n][t] = X[b][t][n]  (f16)
__global__ void k_transpose_X(const float* __restrict__ X, _Float16* __restrict__ Xt)
{
    size_t i = (size_t)blockIdx.x * 256 + threadIdx.x;   // b*N*T + n*T + t
    int t = (int)(i & (T_SZ - 1));
    size_t bn = i >> 6;
    int n = (int)(bn & (N_SZ - 1));
    size_t b = bn >> 8;
    Xt[i] = (_Float16)X[(b * T_SZ + t) * N_SZ + n];
}

// W2[n][0:256]=Wih[n], W2[n][256:512]=Whh[n]   (1024 x 512, f16)
__global__ void k_build_encW(const float* __restrict__ Wih, const float* __restrict__ Whh,
                             _Float16* __restrict__ W2)
{
    int i = blockIdx.x * 256 + threadIdx.x;   // n*512 + k
    int k = i & 511;
    int n = i >> 9;
    float v = (k < 256) ? Wih[n * 256 + k] : Whh[n * 256 + (k - 256)];
    W2[i] = (_Float16)v;
}

__global__ void k_add2(const float* __restrict__ a, const float* __restrict__ b,
                       float* __restrict__ o, int n)
{
    int i = blockIdx.x * 256 + threadIdx.x;
    if (i < n) o[i] = a[i] + b[i];
}

// ---------------------------------------------------------------------------
// Encoder input attention: e[b,n]=sum_s tanh(x[b,s]+y[b,n,s])*ve[s];
// softmax over n; wx = alpha * X[b,t,n] -> wxh[:,0:256] (f16)
// ---------------------------------------------------------------------------
__global__ __launch_bounds__(256)
void enc_attn(const float* __restrict__ x, const _Float16* __restrict__ yh,
              const float* __restrict__ ve, const float* __restrict__ X,
              int t, _Float16* __restrict__ wxh)
{
    __shared__ float xs[T_SZ];
    __shared__ float ves[T_SZ];
    __shared__ float sm[256];
    const int b = blockIdx.x;
    const int n = threadIdx.x;
    if (n < T_SZ) { xs[n] = x[b * T_SZ + n]; ves[n] = ve[n]; }
    __syncthreads();
    const _Float16* yrow = yh + ((size_t)b * N_SZ + n) * T_SZ;
    float e = 0.f;
#pragma unroll 8
    for (int s = 0; s < T_SZ; ++s)
        e += tanhf(xs[s] + (float)yrow[s]) * ves[s];
    sm[n] = e; __syncthreads();
    for (int off = 128; off > 0; off >>= 1) {
        if (n < off) sm[n] = fmaxf(sm[n], sm[n + off]);
        __syncthreads();
    }
    float mx = sm[0]; __syncthreads();
    float ex = __expf(e - mx);
    sm[n] = ex; __syncthreads();
    for (int off = 128; off > 0; off >>= 1) {
        if (n < off) sm[n] += sm[n + off];
        __syncthreads();
    }
    float alpha = ex / sm[0];
    float wx = alpha * X[((size_t)b * T_SZ + t) * N_SZ + n];
    wxh[(size_t)b * 512 + n] = (_Float16)wx;
}

// ---------------------------------------------------------------------------
// Encoder LSTM cell elementwise (PyTorch gate order i,f,g,o)
// ---------------------------------------------------------------------------
__global__ __launch_bounds__(256)
void enc_lstm(const float* __restrict__ g, float* __restrict__ c,
              _Float16* __restrict__ hc, _Float16* __restrict__ wxh,
              _Float16* __restrict__ ench, int t)
{
    const int b = blockIdx.x, m = threadIdx.x;
    const size_t gb = (size_t)b * 1024;
    float gi = g[gb + m], gf = g[gb + 256 + m], gg = g[gb + 512 + m], go = g[gb + 768 + m];
    float cold = c[(size_t)b * 256 + m];
    float c2 = sigmf(gf) * cold + sigmf(gi) * tanhf(gg);
    float h2 = sigmf(go) * tanhf(c2);
    c[(size_t)b * 256 + m] = c2;
    hc[(size_t)b * 512 + m] = (_Float16)h2;
    hc[(size_t)b * 512 + 256 + m] = (_Float16)c2;
    wxh[(size_t)b * 512 + 256 + m] = (_Float16)h2;
    ench[((size_t)b * T_SZ + t) * M_SZ + m] = (_Float16)h2;
}

// ---------------------------------------------------------------------------
// Decoder temporal attention: l[b,t']=sum_m tanh(x1[b,m]+y1[b,t',m])*vd[m];
// beta=softmax_t'; c_t[b,m]=sum_t' beta*enc[b,t',m]; y_til=c_t.wt + wt_b
// ---------------------------------------------------------------------------
__global__ __launch_bounds__(256)
void dec_attn(const float* __restrict__ x1, const _Float16* __restrict__ y1h,
              const _Float16* __restrict__ ench, const float* __restrict__ vd,
              const float* __restrict__ wt, const float* __restrict__ wtb,
              _Float16* __restrict__ dcat, float* __restrict__ ytil)
{
    __shared__ float x1s[M_SZ];
    __shared__ float vds[M_SZ];
    __shared__ float lsh[T_SZ];
    __shared__ float betas[T_SZ];
    __shared__ float red[256];
    const int b = blockIdx.x, tid = threadIdx.x;
    x1s[tid] = x1[(size_t)b * M_SZ + tid];
    vds[tid] = vd[tid];
    __syncthreads();
    if (tid < T_SZ) {
        const _Float16* yrow = y1h + ((size_t)b * T_SZ + tid) * M_SZ;
        float l = 0.f;
        for (int m = 0; m < M_SZ; ++m)
            l += tanhf(x1s[m] + (float)yrow[m]) * vds[m];
        lsh[tid] = l;
    }
    __syncthreads();
    if (tid == 0) {
        float mx = lsh[0];
        for (int i = 1; i < T_SZ; ++i) mx = fmaxf(mx, lsh[i]);
        float s = 0.f;
        for (int i = 0; i < T_SZ; ++i) { betas[i] = __expf(lsh[i] - mx); s += betas[i]; }
        float inv = 1.f / s;
        for (int i = 0; i < T_SZ; ++i) betas[i] *= inv;
    }
    __syncthreads();
    float ct = 0.f;
    const _Float16* eb = ench + (size_t)b * T_SZ * M_SZ + tid;
#pragma unroll 8
    for (int tp = 0; tp < T_SZ; ++tp)
        ct += betas[tp] * (float)eb[(size_t)tp * M_SZ];
    dcat[(size_t)b * 512 + 256 + tid] = (_Float16)ct;
    red[tid] = ct * wt[tid];
    __syncthreads();
    for (int off = 128; off > 0; off >>= 1) {
        if (tid < off) red[tid] += red[tid + off];
        __syncthreads();
    }
    if (tid == 0) ytil[b] = red[0] + wtb[0];
}

// ---------------------------------------------------------------------------
// Decoder LSTM cell: gates = d@Whh^T + bias (from GEMM) + y_til * dec_Wih
// ---------------------------------------------------------------------------
__global__ __launch_bounds__(256)
void dec_lstm(const float* __restrict__ g, const float* __restrict__ wih,
              const float* __restrict__ ytil, float* __restrict__ s,
              _Float16* __restrict__ dsb, _Float16* __restrict__ dcat)
{
    const int b = blockIdx.x, m = threadIdx.x;
    const size_t gb = (size_t)b * 1024;
    const float yt = ytil[b];
    float gi = g[gb + m]       + yt * wih[m];
    float gf = g[gb + 256 + m] + yt * wih[256 + m];
    float gg = g[gb + 512 + m] + yt * wih[512 + m];
    float go = g[gb + 768 + m] + yt * wih[768 + m];
    float sold = s[(size_t)b * 256 + m];
    float s2 = sigmf(gf) * sold + sigmf(gi) * tanhf(gg);
    float d2 = sigmf(go) * tanhf(s2);
    s[(size_t)b * 256 + m] = s2;
    dsb[(size_t)b * 512 + m] = (_Float16)d2;
    dsb[(size_t)b * 512 + 256 + m] = (_Float16)s2;
    dcat[(size_t)b * 512 + m] = (_Float16)d2;
}

// out[b] = dot(tmp[b,:], vy) + vy_b
__global__ __launch_bounds__(256)
void final_out(const float* __restrict__ tmp, const float* __restrict__ vy,
               const float* __restrict__ vyb, float* __restrict__ out)
{
    __shared__ float red[256];
    const int b = blockIdx.x, tid = threadIdx.x;
    red[tid] = tmp[(size_t)b * 256 + tid] * vy[tid];
    __syncthreads();
    for (int off = 128; off > 0; off >>= 1) {
        if (tid < off) red[tid] += red[tid + off];
        __syncthreads();
    }
    if (tid == 0) out[b] = red[0] + vyb[0];
}

// ---------------------------------------------------------------------------
extern "C" void kernel_launch(void* const* d_in, const int* in_sizes, int n_in,
                              void* d_out, int out_size, void* d_ws, size_t ws_size,
                              hipStream_t stream)
{
    (void)in_sizes; (void)n_in; (void)out_size; (void)ws_size;
    const float* X    = (const float*)d_in[0];
    const float* eWih = (const float*)d_in[1];
    const float* eWhh = (const float*)d_in[2];
    const float* ebih = (const float*)d_in[3];
    const float* ebhh = (const float*)d_in[4];
    const float* We_w = (const float*)d_in[5];
    const float* We_b = (const float*)d_in[6];
    const float* Ue_w = (const float*)d_in[7];
    const float* ve_w = (const float*)d_in[8];
    const float* dWih = (const float*)d_in[9];
    const float* dWhh = (const float*)d_in[10];
    const float* dbih = (const float*)d_in[11];
    const float* dbhh = (const float*)d_in[12];
    const float* Wd_w = (const float*)d_in[13];
    const float* Wd_b = (const float*)d_in[14];
    const float* Ud_w = (const float*)d_in[15];
    const float* vd_w = (const float*)d_in[16];
    const float* wt_w = (const float*)d_in[17];
    const float* wt_b = (const float*)d_in[18];
    const float* Wy_w = (const float*)d_in[19];
    const float* Wy_b = (const float*)d_in[20];
    const float* vy_w = (const float*)d_in[21];
    const float* vy_b = (const float*)d_in[22];
    float* out = (float*)d_out;

    char* p = (char*)d_ws;
    auto carve = [&](size_t bytes) -> char* {
        char* r = p; p += (bytes + 255) & ~(size_t)255; return r;
    };
    _Float16* Xt   = (_Float16*)carve((size_t)B_SZ * N_SZ * T_SZ * 2);
    _Float16* yh   = (_Float16*)carve((size_t)B_SZ * N_SZ * T_SZ * 2);
    _Float16* y1h  = (_Float16*)carve((size_t)B_SZ * T_SZ * M_SZ * 2);
    _Float16* ench = (_Float16*)carve((size_t)B_SZ * T_SZ * M_SZ * 2);
    float*    xbuf = (float*)carve((size_t)B_SZ * T_SZ * 4);
    float*    gbuf = (float*)carve((size_t)B_SZ * 1024 * 4);
    _Float16* hc   = (_Float16*)carve((size_t)B_SZ * 512 * 2);
    _Float16* wxh  = (_Float16*)carve((size_t)B_SZ * 512 * 2);
    float*    cbuf = (float*)carve((size_t)B_SZ * 256 * 4);
    float*    x1b  = (float*)carve((size_t)B_SZ * 256 * 4);
    _Float16* dsb  = (_Float16*)carve((size_t)B_SZ * 512 * 2);
    _Float16* dcat = (_Float16*)carve((size_t)B_SZ * 512 * 2);
    float*    sbuf = (float*)carve((size_t)B_SZ * 256 * 4);
    float*    ytil = (float*)carve((size_t)B_SZ * 4);
    float*    tmpb = (float*)carve((size_t)B_SZ * 256 * 4);
    _Float16* WeH  = (_Float16*)carve((size_t)64 * 512 * 2);
    _Float16* W2e  = (_Float16*)carve((size_t)1024 * 512 * 2);
    _Float16* UeH  = (_Float16*)carve((size_t)64 * 64 * 2);
    _Float16* UdH  = (_Float16*)carve((size_t)256 * 256 * 2);
    _Float16* WdH  = (_Float16*)carve((size_t)256 * 512 * 2);
    _Float16* WhdH = (_Float16*)carve((size_t)1024 * 256 * 2);
    _Float16* WyH  = (_Float16*)carve((size_t)256 * 512 * 2);
    float*    benc = (float*)carve(1024 * 4);
    float*    bdec = (float*)carve(1024 * 4);

    // ---- weight prep (f32 -> f16, concat, bias sums) ----
    k_f32_to_f16<<<(64 * 512) / 256, 256, 0, stream>>>(We_w, WeH, 64 * 512);
    k_build_encW<<<(1024 * 512) / 256, 256, 0, stream>>>(eWih, eWhh, W2e);
    k_f32_to_f16<<<(64 * 64) / 256, 256, 0, stream>>>(Ue_w, UeH, 64 * 64);
    k_f32_to_f16<<<(256 * 256) / 256, 256, 0, stream>>>(Ud_w, UdH, 256 * 256);
    k_f32_to_f16<<<(256 * 512) / 256, 256, 0, stream>>>(Wd_w, WdH, 256 * 512);
    k_f32_to_f16<<<(1024 * 256) / 256, 256, 0, stream>>>(dWhh, WhdH, 1024 * 256);
    k_f32_to_f16<<<(256 * 512) / 256, 256, 0, stream>>>(Wy_w, WyH, 256 * 512);
    k_add2<<<4, 256, 0, stream>>>(ebih, ebhh, benc, 1024);
    k_add2<<<4, 256, 0, stream>>>(dbih, dbhh, bdec, 1024);
    k_transpose_X<<<(B_SZ * N_SZ * T_SZ) / 256, 256, 0, stream>>>(X, Xt);

    // ---- zero-init recurrent state ----
    k_zero_u32<<<2048, 256, 0, stream>>>((uint32_t*)hc, B_SZ * 512 / 2);
    k_zero_u32<<<2048, 256, 0, stream>>>((uint32_t*)wxh, B_SZ * 512 / 2);
    k_zero_u32<<<2048, 256, 0, stream>>>((uint32_t*)cbuf, B_SZ * 256);
    k_zero_u32<<<2048, 256, 0, stream>>>((uint32_t*)dsb, B_SZ * 512 / 2);
    k_zero_u32<<<2048, 256, 0, stream>>>((uint32_t*)sbuf, B_SZ * 256);

    // ---- y[b,n,s] = sum_t Xt[b,n,t] * Ue[s,t]  (M=B*N, N=64, K=64) ----
    {
        dim3 g(1, (B_SZ * N_SZ) / 128);
        gemm_wmma_f16<<<g, 128, 0, stream>>>(Xt, 64, UeH, 64, nullptr,
                                             nullptr, yh, 64, 64);
    }

    // ---- encoder scan ----
    for (int t = 0; t < T_SZ; ++t) {
        { dim3 g(1, B_SZ / 128);
          gemm_wmma_f16<<<g, 128, 0, stream>>>(hc, 512, WeH, 512, We_b,
                                               xbuf, nullptr, 64, 512); }
        enc_attn<<<B_SZ, 256, 0, stream>>>(xbuf, yh, ve_w, X, t, wxh);
        { dim3 g(1024 / 64, B_SZ / 128);
          gemm_wmma_f16<<<g, 128, 0, stream>>>(wxh, 512, W2e, 512, benc,
                                               gbuf, nullptr, 1024, 512); }
        enc_lstm<<<B_SZ, 256, 0, stream>>>(gbuf, cbuf, hc, wxh, ench, t);
    }

    // ---- y1 = encoded @ Ud^T  (M=B*T, N=256, K=256) ----
    {
        dim3 g(256 / 64, (B_SZ * T_SZ) / 128);
        gemm_wmma_f16<<<g, 128, 0, stream>>>(ench, 256, UdH, 256, nullptr,
                                             nullptr, y1h, 256, 256);
    }

    // ---- decoder scan ----
    for (int t = 0; t < T_SZ; ++t) {
        { dim3 g(256 / 64, B_SZ / 128);
          gemm_wmma_f16<<<g, 128, 0, stream>>>(dsb, 512, WdH, 512, Wd_b,
                                               x1b, nullptr, 256, 512); }
        dec_attn<<<B_SZ, 256, 0, stream>>>(x1b, y1h, ench, vd_w, wt_w, wt_b,
                                           dcat, ytil);
        { dim3 g(1024 / 64, B_SZ / 128);  // gates = d @ dWhh^T + (bih+bhh)
          gemm_wmma_f16<<<g, 128, 0, stream>>>(dsb, 512, WhdH, 256, bdec,
                                               gbuf, nullptr, 1024, 256); }
        dec_lstm<<<B_SZ, 256, 0, stream>>>(gbuf, dWih, ytil, sbuf, dsb, dcat);
    }

    // ---- final: tmp = [dT|cT] @ Wy^T + Wy_b ; out = tmp @ vy^T + vy_b ----
    {
        dim3 g(256 / 64, B_SZ / 128);
        gemm_wmma_f16<<<g, 128, 0, stream>>>(dcat, 512, WyH, 512, Wy_b,
                                             tmpb, nullptr, 256, 512);
    }
    final_out<<<B_SZ, 256, 0, stream>>>(tmpb, vy_w, vy_b, out);
}
